// convolution_EQ_66967130079931
// MI455X (gfx1250) — matrix operand order repaired and checked by reading, not compiled, not run on hardware
//
#include <hip/hip_runtime.h>
#include <hip/hip_bf16.h>

// ---------------- problem constants ----------------
#define BATCH   16
#define T_OUT   176400
#define HOP     88
#define NFFT    441
#define NBAND   65
#define IRLEN   129          // 2*65-1
#define LF      568          // reference irfft default length used for OLA
#define NFRAMES 2004
#define NROWS   (BATCH * NFRAMES)   // 32064
#define PAD     176          // (441-88)/2

// GEMM padding: K 65 -> 96 (3 x 32), N 129 -> 144 (9 x 16)
#define KP      96
#define NP      144
#define NT_COL  9
#define NT_ROW  (NROWS / 16)        // 2004
#define NTILES  (NT_ROW * NT_COL)   // 18036

#define TWO_PI 6.28318530717958647692f

typedef __attribute__((ext_vector_type(16))) _Float16 v16h;
typedef __attribute__((ext_vector_type(8)))  _Float16 v8h;
typedef __attribute__((ext_vector_type(2)))  _Float16 v2h;
typedef __attribute__((ext_vector_type(8)))  float    v8f;

// ---------------------------------------------------------------------------
// Kernel 1: build Mt (f16, n-major: Mt[n*KP + k], zero-padded) and hann441.
//   M[n][k] = hann129[n] * (k==0?1:2)/129 * cos(2*pi*k*((n+65) mod 129)/129)
// ---------------------------------------------------------------------------
__global__ __launch_bounds__(256) void eq_init(_Float16* __restrict__ Mt,
                                               float* __restrict__ hann441) {
    int i = blockIdx.x * blockDim.x + threadIdx.x;
    if (i < NP * KP) {
        int n = i / KP;
        int k = i % KP;
        float v = 0.0f;
        if (k < NBAND && n < IRLEN) {
            int m = (n + NBAND) % IRLEN;                  // roll by nb-1 = 64
            float c = __cosf(TWO_PI * (float)(k * m) / (float)IRLEN);
            float h = 0.5f - 0.5f * __cosf(TWO_PI * (float)n / (float)IRLEN);
            float s = (k == 0) ? 1.0f : 2.0f;
            v = h * s * c / (float)IRLEN;
        }
        Mt[i] = (_Float16)v;
    }
    if (i < NFFT) {
        hann441[i] = 0.5f - 0.5f * __cosf(TWO_PI * (float)i / (float)NFFT);
    }
}

// ---------------------------------------------------------------------------
// Kernel 2: pack EQ_FR f32 (R x 65) -> Zh f16 (R x 96, zero-padded).
// One thread per f16 pair; branch-free wide stores, coalesced f32 reads.
// ---------------------------------------------------------------------------
__global__ __launch_bounds__(256) void eq_pack_z(const float* __restrict__ Z,
                                                 _Float16* __restrict__ Zh) {
    long i = (long)blockIdx.x * blockDim.x + threadIdx.x;   // pair index
    if (i >= (long)NROWS * (KP / 2)) return;
    int r = (int)(i / (KP / 2));
    int k = 2 * (int)(i % (KP / 2));
    float v0 = (k     < NBAND) ? Z[(size_t)r * NBAND + k]     : 0.0f;
    float v1 = (k + 1 < NBAND) ? Z[(size_t)r * NBAND + k + 1] : 0.0f;
    v2h p = { (_Float16)v0, (_Float16)v1 };
    *(v2h*)(Zh + (size_t)r * KP + k) = p;
}

// ---------------------------------------------------------------------------
// Kernel 3: IR[32064 x 144] = Zh[32064 x 96] @ Mt^T  via WMMA f16->f32.
// One wave per 16x16 tile, 3 K-chunks of 32, vectorized fragment loads.
// A lane layout: halves 0-15 -> K = base+16*hi+0..15, row = lane%16.
// B lane layout: halves 0-15 -> K = base+16*hi+0..15, column = lane%16.
// D: vgpr v -> row v+8*hi, col lane%16.
// ---------------------------------------------------------------------------
__global__ __launch_bounds__(256) void eq_ir_gemm(const _Float16* __restrict__ Zh,
                                                  const _Float16* __restrict__ Mt,
                                                  float* __restrict__ IR) {
    const int wave = (int)(threadIdx.x >> 5);
    const int tile = (int)blockIdx.x * 8 + wave;
    if (tile >= NTILES) return;                // uniform per wave: EXEC stays all-1
    const int rt = tile / NT_COL;
    const int nt = tile % NT_COL;
    const int lane = (int)(threadIdx.x & 31);
    const int hi = lane >> 4;
    const int lo = lane & 15;

    const _Float16* __restrict__ zrow = Zh + (size_t)(rt * 16 + lo) * KP;
    const _Float16* __restrict__ mrow = Mt + (size_t)(nt * 16 + lo) * KP;

    v8f acc = {};
#pragma unroll
    for (int kc = 0; kc < 3; ++kc) {
        const int base = kc * 32;
        v16h a = *(const v16h*)(zrow + base + 16 * hi);       // 32B aligned
        v16h b = *(const v16h*)(mrow + base + 16 * hi);       // 32B aligned
        acc = __builtin_amdgcn_wmma_f32_16x16x32_f16(
            /*neg_a=*/false, a, /*neg_b=*/false, b,
            /*c_mod=*/(short)0, acc, /*reuse_a=*/false, /*reuse_b=*/false);
    }
#pragma unroll
    for (int vg = 0; vg < 8; ++vg) {
        int m = vg + 8 * hi;
        IR[(size_t)(rt * 16 + m) * NP + nt * 16 + lo] = acc[vg];
    }
}

// ---------------------------------------------------------------------------
// Kernel 4: fused windowed-conv + overlap-add, gather form (no atomics):
//   out[b,s] = sum_f sum_t IR[b,f,t] * wf_f[s-176-t]
// where wf_f[g] = hann441[g-(88f-176)] * audio[g] (zero outside window).
// Pre-windowed frames live in LDS; the 129-tap inner loop has a fixed trip
// count, broadcast ir[t], stride-1 wf -> conflict-free, divergence-free.
// ---------------------------------------------------------------------------
#define BLK    256
#define MAXFR  12
#define AWIN   (BLK + IRLEN - 1)   // 384 audio samples per block

__global__ __launch_bounds__(BLK) void eq_conv_ola(const float* __restrict__ audio,
                                                   const float* __restrict__ IR,
                                                   const float* __restrict__ hann441,
                                                   float* __restrict__ out) {
    __shared__ float s_audio[AWIN];
    __shared__ float s_hann[NFFT];
    __shared__ float s_ir[MAXFR][IRLEN + 1];
    __shared__ float s_wf[MAXFR][AWIN];

    const int b   = (int)blockIdx.y;
    const int s0  = (int)blockIdx.x * BLK;
    const int tid = (int)threadIdx.x;

    // audio[b, s-176-t] for s in [s0, s0+255], t in [0,128] -> [s0-304, s0+79]
    const int a0 = s0 - (PAD + IRLEN - 1);     // s0 - 304
    for (int i = tid; i < AWIN; i += BLK) {
        int gi = a0 + i;
        s_audio[i] = (gi >= 0 && gi < T_OUT) ? audio[(size_t)b * T_OUT + gi] : 0.0f;
        // prefetch the next block's audio window (global_prefetch_b8)
        int pg = gi + BLK;
        if (pg >= 0 && pg < T_OUT) __builtin_prefetch(audio + (size_t)b * T_OUT + pg, 0, 1);
    }
    for (int i = tid; i < NFFT; i += BLK) s_hann[i] = hann441[i];

    // contributing frames: ceil((s0-567)/88) <= f <= floor((s0+255)/88)
    int flo = s0 - (LF - 1);
    flo = (flo <= 0) ? 0 : (flo + HOP - 1) / HOP;
    int fhi = (s0 + BLK - 1) / HOP;
    if (fhi > NFRAMES - 1) fhi = NFRAMES - 1;
    int nf = fhi - flo + 1;
    if (nf > MAXFR) nf = MAXFR;
    if (nf < 0) nf = 0;

    for (int i = tid; i < nf * IRLEN; i += BLK) {
        int fi = i / IRLEN, t = i % IRLEN;
        s_ir[fi][t] = IR[(size_t)(b * NFRAMES + flo + fi) * NP + t];
    }
    __syncthreads();

    // pre-windowed frames over this block's audio window
    for (int i = tid; i < nf * AWIN; i += BLK) {
        int fi = i / AWIN, j = i % AWIN;
        int ih = (a0 + j) - HOP * (flo + fi) + PAD;   // hann441 index
        s_wf[fi][j] = (ih >= 0 && ih < NFFT) ? s_hann[ih] * s_audio[j] : 0.0f;
    }
    __syncthreads();

    const int s = s0 + tid;
    if (s >= T_OUT) return;

    const int abase = tid + (IRLEN - 1);       // (s-176-t) - a0 = tid + 128 - t
    float acc = 0.0f;
    for (int fi = 0; fi < nf; ++fi) {
        const int n = s - HOP * (flo + fi);    // position inside frame's conv output
        if (n < 0 || n > LF - 1) continue;     // y-length truncation (568)
        const float* __restrict__ irp = s_ir[fi];
        const float* __restrict__ wfp = s_wf[fi];
#pragma unroll 8
        for (int t = 0; t < IRLEN; ++t) {
            acc = fmaf(irp[t], wfp[abase - t], acc);   // OOB taps hit wf==0
        }
    }
    out[(size_t)b * T_OUT + s] = acc;
}

// ---------------------------------------------------------------------------
extern "C" void kernel_launch(void* const* d_in, const int* in_sizes, int n_in,
                              void* d_out, int out_size, void* d_ws, size_t ws_size,
                              hipStream_t stream) {
    const float* audio = (const float*)d_in[0];   // (16, 176400) f32
    const float* eqfr  = (const float*)d_in[1];   // (16, 2004, 65) f32
    float* out = (float*)d_out;                   // (16, 176400) f32

    char* ws = (char*)d_ws;
    _Float16* Mt   = (_Float16*)ws;                        // 144*96 f16 = 27,648 B
    float*    hann = (float*)(ws + 27648);                 // 441 f32    =  1,764 B
    _Float16* Zh   = (_Float16*)(ws + 32768);              // 32064*96 f16 = 6,156,288 B
    float*    IR   = (float*)(ws + 32768 + 6156288);       // 32064*144 f32 = 18.47 MB

    // 1) synthesis matrix (transposed) + analysis window
    eq_init<<<(NP * KP + 255) / 256, 256, 0, stream>>>(Mt, hann);

    // 2) pack/pad EQ bins to f16
    eq_pack_z<<<(int)(((long)NROWS * (KP / 2) + 255) / 256), 256, 0, stream>>>(eqfr, Zh);

    // 3) IR bank GEMM on the matrix pipes (v_wmma_f32_16x16x32_f16)
    eq_ir_gemm<<<(NTILES + 7) / 8, 256, 0, stream>>>(Zh, Mt, IR);

    // 4) fused windowed convolution + overlap-add (gather, one write per output)
    dim3 grid((T_OUT + BLK - 1) / BLK, BATCH);
    eq_conv_ola<<<grid, BLK, 0, stream>>>(audio, IR, hann, out);
}
